// ULOSD_21912923144745
// MI455X (gfx1250) — matrix operand order repaired and checked by reading, not compiled, not run on hardware
//
#include <hip/hip_runtime.h>

typedef __attribute__((ext_vector_type(16))) _Float16 v16h;
typedef __attribute__((ext_vector_type(8)))  float    v8f;
typedef __attribute__((ext_vector_type(4)))  unsigned int u32x4;

union Frag16 { v16h v; u32x4 q[2]; };

#define ACT_ID    0
#define ACT_LEAKY 1
#define ACT_RELU  2
#define ACT_SP    3

#define OUT_F16_NHWC 0
#define OUT_F32_NHWC 1
#define OUT_F32_NCHW 2

template<int ACT> __device__ __forceinline__ float actf(float v) {
  if constexpr (ACT == ACT_LEAKY) return v > 0.f ? v : 0.2f * v;
  else if constexpr (ACT == ACT_RELU) return v > 0.f ? v : 0.f;
  else if constexpr (ACT == ACT_SP) return fmaxf(v, 0.f) + log1pf(expf(-fabsf(v)));
  else return v;
}

// ---------------------------------------------------------------------------
// Implicit-GEMM conv / transpose-conv on v_wmma_f32_16x16x32_f16.
// One wave: 16 x-positions (M) x NT*16 output channels (N).  Fully unrolled
// K loop; boundary handling is a branch-free pointer select onto a zero page,
// so global_load_b128s can be hoisted freely across WMMAs by the scheduler.
// ---------------------------------------------------------------------------
template<int Cin,int Cout,int Hin,int Win,int Hout,int Wout,int STRIDE,bool TCONV,
         int ACT,int OUTMODE,int NT>
__global__ __launch_bounds__(256) void convWmma(
    const _Float16* __restrict__ act, const _Float16* __restrict__ wpk,
    const float* __restrict__ bias, void* __restrict__ outp,
    const _Float16* __restrict__ zpage, int nTiles)
{
  constexpr int nOb    = Cout / 16;
  constexpr int nObW   = nOb / NT;          // ob-groups per tile column
  constexpr int tilesX = Wout / 16;
  constexpr int CB     = (Cin >= 32) ? (Cin / 32) : 1;

  const int wid = blockIdx.x * (blockDim.x >> 5) + (threadIdx.x >> 5);
  if (wid >= nTiles) return;
  const int lane = threadIdx.x & 31;
  const int n    = lane & 15;
  const int grp8 = (lane & 16) >> 1;        // 0 or 8

  int obw = wid % nObW;
  int t1  = wid / nObW;
  int xt  = t1 % tilesX;
  int t2  = t1 / tilesX;
  int y   = t2 % Hout;
  int b   = t2 / Hout;
  const int obBase = obw * NT;

  const int xm = xt * 16 + n;               // this lane's A-row (spatial x)

  v8f c[NT];
  #pragma unroll
  for (int nt = 0; nt < NT; ++nt) c[nt] = (v8f){0.f,0.f,0.f,0.f,0.f,0.f,0.f,0.f};

  // Base pointer for B fragments of this wave (immediate offsets per k-step).
  const _Float16* __restrict__ wB = wpk + ((size_t)(obBase * 32 + lane)) * 16;

  if constexpr (Cin >= 32) {
    #pragma unroll
    for (int tap = 0; tap < 9; ++tap) {
      const int dy = tap / 3, dx = tap - (tap / 3) * 3;   // constants after unroll
      int iy, ix; bool ok;
      if constexpr (!TCONV) {
        constexpr int PAD = (STRIDE == 1) ? 1 : 0;        // JAX SAME: s1->(1,1), s2->(0,1)
        iy = y  * STRIDE + dy - PAD;
        ix = xm * STRIDE + dx - PAD;
        ok = ((unsigned)iy < (unsigned)Hin) && ((unsigned)ix < (unsigned)Win);
      } else {                                            // tconv s=2,k=3,SAME: pad_a=2
        int d0 = y  + dy - 2;
        int d1 = xm + dx - 2;
        iy = d0 >> 1; ix = d1 >> 1;
        ok = (d0 >= 0) && (d1 >= 0) && !(d0 & 1) && !(d1 & 1) &&
             (iy < Hin) && (ix < Win);
      }
      size_t off = ok ? ((size_t)((b * Hin + iy) * Win + ix) * (size_t)Cin
                         + (size_t)grp8)
                      : (size_t)0;
      const _Float16* __restrict__ pA = ok ? (act + off) : zpage;  // zero page: 512 B

      #pragma unroll
      for (int cb = 0; cb < CB; ++cb) {
        Frag16 a;
        a.q[0] = *(const u32x4*)(pA + cb * 32);
        a.q[1] = *(const u32x4*)(pA + cb * 32 + 16);
        const int s = tap * CB + cb;
        #pragma unroll
        for (int nt = 0; nt < NT; ++nt) {
          Frag16 bf;
          const _Float16* pB = wB + ((size_t)s * nOb + nt) * 512;
          bf.q[0] = *(const u32x4*)(pB);
          bf.q[1] = *(const u32x4*)(pB + 8);
          c[nt] = __builtin_amdgcn_wmma_f32_16x16x32_f16(false, a.v, false, bf.v,
                                                         (short)0, c[nt], false, false);
        }
      }
    }
  } else {
    // Cin == 16: two taps per K=32 WMMA (lone 9th tap has zeroed upper-K weights)
    #pragma unroll
    for (int s = 0; s < 5; ++s) {
      int taps[2];
      if (s < 4) { taps[0] = 2*s; taps[1] = 2*s + 1; }
      else       { taps[0] = 8;   taps[1] = 8; }
      Frag16 a;
      #pragma unroll
      for (int ch = 0; ch < 2; ++ch) {
        const int tap = taps[ch];
        const int dy = tap / 3, dx = tap - (tap / 3) * 3;
        int iy, ix; bool ok;
        if constexpr (!TCONV) {
          constexpr int PAD = (STRIDE == 1) ? 1 : 0;
          iy = y  * STRIDE + dy - PAD;
          ix = xm * STRIDE + dx - PAD;
          ok = ((unsigned)iy < (unsigned)Hin) && ((unsigned)ix < (unsigned)Win);
        } else {
          int d0 = y  + dy - 2;
          int d1 = xm + dx - 2;
          iy = d0 >> 1; ix = d1 >> 1;
          ok = (d0 >= 0) && (d1 >= 0) && !(d0 & 1) && !(d1 & 1) &&
               (iy < Hin) && (ix < Win);
        }
        size_t off = ok ? ((size_t)((b * Hin + iy) * Win + ix) * 16u + (size_t)grp8)
                        : (size_t)0;
        const _Float16* pA = ok ? (act + off) : zpage;
        a.q[ch] = *(const u32x4*)pA;
      }
      #pragma unroll
      for (int nt = 0; nt < NT; ++nt) {
        Frag16 bf;
        const _Float16* pB = wB + ((size_t)s * nOb + nt) * 512;
        bf.q[0] = *(const u32x4*)(pB);
        bf.q[1] = *(const u32x4*)(pB + 8);
        c[nt] = __builtin_amdgcn_wmma_f32_16x16x32_f16(false, a.v, false, bf.v,
                                                       (short)0, c[nt], false, false);
      }
    }
  }

  #pragma unroll
  for (int nt = 0; nt < NT; ++nt) {
    const int o = (obBase + nt) * 16 + n;
    const float bv = bias[o];
    #pragma unroll
    for (int r = 0; r < 8; ++r) {
      float v = actf<ACT>(c[nt][r] + bv);
      int M  = r + grp8;                     // C/D layout: row M = r + 8*(lane>=16)
      int xo = xt * 16 + M;
      if constexpr (OUTMODE == OUT_F16_NHWC) {
        ((_Float16*)outp)[((size_t)((b * Hout + y) * Wout + xo)) * Cout + o] = (_Float16)v;
      } else if constexpr (OUTMODE == OUT_F32_NHWC) {
        ((float*)outp)[((size_t)((b * Hout + y) * Wout + xo)) * Cout + o] = v;
      } else {
        ((float*)outp)[(((size_t)(b * Cout + o)) * Hout + y) * Wout + xo] = v;
      }
    }
  }
}

// ---------------------------------------------------------------------------
// Pre-pack OIHW f32 weights into per-(kstep, ob) B-fragment lane order (f16).
// B layout (32x16 f16): lane -> column N = lane&15, K = (e&7) + 8*(lane>=16) + 16*(e>=8).
// ---------------------------------------------------------------------------
__global__ void prepackKernel(const float* __restrict__ w, _Float16* __restrict__ pk,
                              int Cin, int Cout, int total)
{
  int t = blockIdx.x * blockDim.x + threadIdx.x;
  if (t >= total) return;
  int e    = t & 15;
  int lane = (t >> 4) & 31;
  int rest = t >> 9;                 // = s*nOb + ob
  int nOb  = Cout >> 4;
  int ob   = rest % nOb;
  int s    = rest / nOb;
  int k    = (e & 7) + ((lane & 16) >> 1) + ((e & 8) << 1);        // K index 0..31
  int o    = ob * 16 + (lane & 15);
  int tap = 0, cch = 0;
  bool zero = false;
  if (Cin >= 32) {
    int CBv = Cin >> 5;
    tap = s / CBv;
    cch = (s - tap * CBv) * 32 + k;
  } else {                           // Cin == 16, paired taps
    if (k < 16)      { tap = (s < 4) ? 2*s : 8; cch = k; }
    else if (s < 4)  { tap = 2*s + 1; cch = k - 16; }
    else             { zero = true; }
  }
  float val = 0.f;
  if (!zero) {
    int dy = tap / 3, dx = tap - dy * 3;
    val = w[(((o * Cin) + cch) * 3 + dy) * 3 + dx];
  }
  pk[t] = (_Float16)val;
}

__global__ void zeroKernel(_Float16* __restrict__ p) {  // 512-byte zero page
  p[threadIdx.x] = (_Float16)0.f;
}

// ---------------------------------------------------------------------------
// enc0: 3->32, stride 1, identity.  Cheap (0.45 GMAC) -> plain VALU f32 conv.
// Reads NCHW f32 input, writes NHWC f16.  Exactly 32*128*128*32 threads.
// ---------------------------------------------------------------------------
__global__ __launch_bounds__(256) void enc0Kernel(
    const float* __restrict__ x, const float* __restrict__ w,
    const float* __restrict__ bias, _Float16* __restrict__ out)
{
  unsigned t = blockIdx.x * 256u + threadIdx.x;
  int o  = t & 31;
  int xx = (t >> 5) & 127;
  int y  = (t >> 12) & 127;
  int b  = t >> 19;
  float acc = bias[o];
  #pragma unroll
  for (int cc = 0; cc < 3; ++cc)
    #pragma unroll
    for (int dy = 0; dy < 3; ++dy) {
      int iy = y + dy - 1;
      if ((unsigned)iy >= 128u) continue;
      #pragma unroll
      for (int dx = 0; dx < 3; ++dx) {
        int ix = xx + dx - 1;
        if ((unsigned)ix >= 128u) continue;
        acc += x[(((size_t)(b * 3 + cc) * 128) + iy) * 128 + ix] *
               w[((o * 3 + cc) * 3 + dy) * 3 + dx];
      }
    }
  out[t] = (_Float16)acc;
}

// ---------------------------------------------------------------------------
// Keypoint middle section (fp32): mu, last-map denom, soft-argmax coords,
// gaussian blob maps.  R is f32 NHWC (32,16,16,128); maps out f16 NHWC.
// ---------------------------------------------------------------------------
__global__ void keypointKernel(const float* __restrict__ R, _Float16* __restrict__ maps)
{
  const int b = blockIdx.x;
  const int k = threadIdx.x;                 // 0..127
  const float* Rb = R + (size_t)b * 256 * 128;
  float s = 0.f, sy = 0.f, sx = 0.f;
  for (int p = 0; p < 256; ++p) {
    float v = Rb[p * 128 + k];
    s  += v;
    sy += v * (float)(p >> 4);               // row (h) coordinate
    sx += v * (float)(p & 15);               // col (w) coordinate
  }
  __shared__ float den;
  if (k == 127) den = s;                     // sum of ORIGINAL last map
  __syncthreads();
  float d  = den;
  float mu = s * (1.f / 256.f);
  float c0 = sy / d;
  float c1 = sx / d;
  _Float16* Mb = maps + (size_t)b * 256 * 128;
  for (int p = 0; p < 256; ++p) {
    float du = (float)(p >> 4) - c0;
    float dv = (float)(p & 15) - c1;
    Mb[p * 128 + k] = (_Float16)(mu * expf(-0.5f * (du * du + dv * dv)));
  }
}

// ---------------------------------------------------------------------------
extern "C" void kernel_launch(void* const* d_in, const int* in_sizes, int n_in,
                              void* d_out, int out_size, void* d_ws, size_t ws_size,
                              hipStream_t stream)
{
  (void)in_sizes; (void)n_in; (void)out_size; (void)ws_size;
  const float* x = (const float*)d_in[0];

  char* ws = (char*)d_ws;
  _Float16* bufA    = (_Float16*)(ws);                        // 33,554,432 B
  _Float16* bufB    = (_Float16*)(ws + (size_t)33554432);     // 33,554,432 B
  float*    Rbuf    = (float*)   (ws + (size_t)67108864);     //  4,194,304 B
  _Float16* wpkBase = (_Float16*)(ws + (size_t)71303168);     // ~1.16 MB packed weights

  // WMMA layers: enc1..enc6, dec0..dec5  (weight d_in index, Cin, Cout)
  const int cin [12] = {32,32,64, 64,128,128, 128,64,64,32,32,16};
  const int cout[12] = {32,64,64,128,128,128,  64,64,32,32,16,16};
  const int widx[12] = { 3, 5, 7,  9, 11, 13,  15,17,19,21,23,25};
  _Float16* pk[12];
  size_t off = 0;
  for (int i = 0; i < 12; ++i) {
    int NK    = (cin[i] >= 32) ? 9 * (cin[i] / 32) : 5;
    int total = NK * (cout[i] / 16) * 512;
    pk[i] = wpkBase + off;
    off  += (size_t)total;
    prepackKernel<<<(total + 255) / 256, 256, 0, stream>>>(
        (const float*)d_in[widx[i]], pk[i], cin[i], cout[i], total);
  }
  _Float16* zpage = wpkBase + off;                            // 512-byte zero page
  zeroKernel<<<1, 256, 0, stream>>>(zpage);

  // encoder
  enc0Kernel<<<65536, 256, 0, stream>>>(x, (const float*)d_in[1],
                                        (const float*)d_in[2], bufA);
  { int t = 32768; convWmma<32,32,128,128,128,128,1,false,ACT_ID,OUT_F16_NHWC,2>
      <<<(t+7)/8,256,0,stream>>>(bufA, pk[0], (const float*)d_in[4],  bufB, zpage, t); }
  { int t = 16384; convWmma<32,64,128,128, 64, 64,2,false,ACT_LEAKY,OUT_F16_NHWC,2>
      <<<(t+7)/8,256,0,stream>>>(bufB, pk[1], (const float*)d_in[6],  bufA, zpage, t); }
  { int t = 16384; convWmma<64,64, 64, 64, 64, 64,1,false,ACT_LEAKY,OUT_F16_NHWC,2>
      <<<(t+7)/8,256,0,stream>>>(bufA, pk[2], (const float*)d_in[8],  bufB, zpage, t); }
  { int t =  8192; convWmma<64,128,64, 64, 32, 32,2,false,ACT_LEAKY,OUT_F16_NHWC,2>
      <<<(t+7)/8,256,0,stream>>>(bufB, pk[3], (const float*)d_in[10], bufA, zpage, t); }
  { int t =  8192; convWmma<128,128,32,32, 32, 32,1,false,ACT_LEAKY,OUT_F16_NHWC,2>
      <<<(t+7)/8,256,0,stream>>>(bufA, pk[4], (const float*)d_in[12], bufB, zpage, t); }
  { int t =  2048; convWmma<128,128,32,32, 16, 16,2,false,ACT_SP,OUT_F32_NHWC,2>
      <<<(t+7)/8,256,0,stream>>>(bufB, pk[5], (const float*)d_in[14], Rbuf, zpage, t); }

  // keypoints -> gaussian blob maps (into bufA as f16 NHWC)
  keypointKernel<<<32, 128, 0, stream>>>(Rbuf, bufA);

  // decoder
  { int t =  4096; convWmma<128,64,16,16, 32, 32,2,true, ACT_RELU,OUT_F16_NHWC,2>
      <<<(t+7)/8,256,0,stream>>>(bufA, pk[6], (const float*)d_in[16], bufB, zpage, t); }
  { int t =  4096; convWmma<64,64, 32,32, 32, 32,1,false,ACT_RELU,OUT_F16_NHWC,2>
      <<<(t+7)/8,256,0,stream>>>(bufB, pk[7], (const float*)d_in[18], bufA, zpage, t); }
  { int t =  8192; convWmma<64,32, 32,32, 64, 64,2,true, ACT_RELU,OUT_F16_NHWC,2>
      <<<(t+7)/8,256,0,stream>>>(bufA, pk[8], (const float*)d_in[20], bufB, zpage, t); }
  { int t =  8192; convWmma<32,32, 64,64, 64, 64,1,false,ACT_RELU,OUT_F16_NHWC,2>
      <<<(t+7)/8,256,0,stream>>>(bufB, pk[9], (const float*)d_in[22], bufA, zpage, t); }
  { int t = 32768; convWmma<32,16, 64,64,128,128,2,true, ACT_RELU,OUT_F16_NHWC,1>
      <<<(t+7)/8,256,0,stream>>>(bufA, pk[10],(const float*)d_in[24], bufB, zpage, t); }
  { int t = 32768; convWmma<16,16,128,128,128,128,1,false,ACT_RELU,OUT_F32_NCHW,1>
      <<<(t+7)/8,256,0,stream>>>(bufB, pk[11],(const float*)d_in[26], d_out, zpage, t); }
}
